// PolyhedronModel_5351529251221
// MI455X (gfx1250) — matrix operand order repaired and checked by reading, not compiled, not run on hardware
//
#include <hip/hip_runtime.h>
#include <stdint.h>

// ---------------------------------------------------------------------------
// CGConv on MI455X (gfx1250):
//   edge GEMM  : v_wmma_f32_16x16x32_f16, one 16-edge tile per wave32
//   z staging  : global gather -> v_cvt_pk_rtz_f16_f32 -> LDS (A layout)
//   weights    : pre-swizzled into B-fragment lane layout, cached in LDS
//   scatter    : global_atomic_add_f32 (agg is L2-resident, 12.8 MB)
// ---------------------------------------------------------------------------

typedef __attribute__((ext_vector_type(16))) _Float16 v16h;
typedef __attribute__((ext_vector_type(8)))  _Float16 v8h;
typedef __attribute__((ext_vector_type(2)))  __fp16   v2hf;   // cvt_pkrtz result type
typedef __attribute__((ext_vector_type(8)))  float    v8f;

__device__ __forceinline__ uint32_t pack_f16x2(float a, float b) {
    union { v2hf h; uint32_t u; } cvt;
    cvt.h = __builtin_amdgcn_cvt_pkrtz(a, b);
    return cvt.u;
}

#define C_DIM   64
#define DE      32
#define ZK      160          // 2*C + DE
#define KTILES  5            // 160 / 32
#define NTILES  4            // 64 / 16
#define NFRAG   40           // 2 weights * KTILES * NTILES
#define WAVES   4
#define TPW     4            // edge tiles per wave

// ws layout (bytes)
#define OFF_B     0          // 40960 B : swizzled f16 B fragments
#define OFF_GPOOL 49152      // G*6*4 B
#define OFF_AGG   65536      // N*64*4 B

__device__ __forceinline__ float fsig(float x) {
    return 1.0f / (1.0f + __expf(-x));
}
__device__ __forceinline__ float fsoftplus(float x) {
    return (x > 20.0f) ? x : __logf(1.0f + __expf(x));
}

// ---------------------------------------------------------------------------
// Swizzle W_f/W_s [160,64] f32 row-major into WMMA B-fragment layout (f16).
// Fragment f = (w*KTILES + kt)*NTILES + nt, 1024 B each:
//   u32[f*256 + g*128 + lane*4 + j], v = g*4+j holds packed (K=k0, K=k0+1)
//   for column n = nt*16 + (lane&15), k0 = kt*32 + (lane>=16 ? 16:0) + 2v.
// ---------------------------------------------------------------------------
__global__ void prep_weights_kernel(const float* __restrict__ Wf,
                                    const float* __restrict__ Ws,
                                    uint32_t* __restrict__ wsB) {
    int t = blockIdx.x * blockDim.x + threadIdx.x;
    if (t >= NFRAG * 256) return;
    int f   = t >> 8;
    int r   = t & 255;
    int g   = r >> 7;
    int l   = (r >> 2) & 31;
    int j   = r & 3;
    int w   = f / (KTILES * NTILES);
    int rem = f % (KTILES * NTILES);
    int kt  = rem / NTILES;
    int nt  = rem % NTILES;
    int v   = g * 4 + j;
    int n   = nt * 16 + (l & 15);
    int k0  = kt * 32 + ((l >= 16) ? 16 : 0) + 2 * v;
    const float* W = w ? Ws : Wf;
    wsB[t] = pack_f16x2(W[k0 * C_DIM + n], W[(k0 + 1) * C_DIM + n]);
}

// ---------------------------------------------------------------------------
// Edge kernel: each wave handles TPW tiles of 16 edges.
// ---------------------------------------------------------------------------
__launch_bounds__(128)
__global__ void edge_kernel(const float* __restrict__ x,
                            const long long* __restrict__ ei,   // [2,E] src row0, dst row1
                            const float* __restrict__ ea,       // [E,32]
                            const float* __restrict__ bf_,
                            const float* __restrict__ bs_,
                            const uint32_t* __restrict__ wsB,
                            float* __restrict__ agg,
                            int E, int numTiles) {
    __shared__ __align__(16) uint32_t  ldsB[NFRAG * 256];        // 40960 B
    __shared__ __align__(16) _Float16  ldsZ[WAVES][16 * ZK];     // 4*5120 B
    __shared__ __align__(16) int       ldsIdx[WAVES][32];        // [0:15]=dst [16:31]=src

    const int tid = threadIdx.x;

    // cooperative B copy: 2560 uint4
    {
        const uint4* s4 = (const uint4*)wsB;
        uint4*       d4 = (uint4*)ldsB;
        for (int i = tid; i < NFRAG * 64; i += 128) d4[i] = s4[i];
    }
    __syncthreads();

    const int  wv     = tid >> 5;
    const int  l      = tid & 31;
    const int  laneLo = l & 15;
    const bool hiHalf = (l >= 16);

    for (int t = 0; t < TPW; ++t) {
        int tile = (blockIdx.x * WAVES + wv) * TPW + t;
        if (tile >= numTiles) break;
        int e0 = tile * 16;

        // --- stage edge indices: lanes 0-15 -> dst (row 1), 16-31 -> src (row 0)
        {
            long long idx = hiHalf ? ei[(size_t)e0 + laneLo]
                                   : ei[(size_t)E + e0 + laneLo];
            ldsIdx[wv][l] = (int)idx;
        }
        __builtin_amdgcn_wave_barrier();

        // --- gather z = [x[dst] | x[src] | edge_attr] as f16 rows in LDS
        for (int r = 0; r < 16; ++r) {
            int d = ldsIdx[wv][r];
            int s = ldsIdx[wv][16 + r];
            float2 xd = ((const float2*)(x + (long long)d * C_DIM))[l];
            float2 xs = ((const float2*)(x + (long long)s * C_DIM))[l];
            *(uint32_t*)&ldsZ[wv][r * ZK + 2 * l]         = pack_f16x2(xd.x, xd.y);
            *(uint32_t*)&ldsZ[wv][r * ZK + C_DIM + 2 * l] = pack_f16x2(xs.x, xs.y);
            float a = ea[(long long)(e0 + r) * DE + l];
            ldsZ[wv][r * ZK + 2 * C_DIM + l] = (_Float16)a;
        }
        __builtin_amdgcn_wave_barrier();

        // --- WMMA: acc[w][nt] += A(kt) * B(w,kt,nt)
        v8f accF[NTILES], accS[NTILES];
        #pragma unroll
        for (int i = 0; i < NTILES; ++i) { accF[i] = (v8f)0.0f; accS[i] = (v8f)0.0f; }

        const _Float16* zb = &ldsZ[wv][0];
        #pragma unroll
        for (int kt = 0; kt < KTILES; ++kt) {
            // A fragment (contiguous k order, same as B): lane half selects K +0 / +16
            const v8h* ap = (const v8h*)(zb + laneLo * ZK + kt * 32 + (hiHalf ? 16 : 0));
            v8h a0 = ap[0];       // K 16h+0..7
            v8h a1 = ap[1];       // K 16h+8..15 (+16 B)
            v16h A;
            #pragma unroll
            for (int i = 0; i < 8; ++i) { A[i] = a0[i]; A[8 + i] = a1[i]; }

            #pragma unroll
            for (int w2 = 0; w2 < 2; ++w2) {
                #pragma unroll
                for (int nt = 0; nt < NTILES; ++nt) {
                    int f = (w2 * KTILES + kt) * NTILES + nt;
                    const uint32_t* fb = &ldsB[f * 256 + l * 4];
                    union { uint4 q; v8h h; } b0, b1;
                    b0.q = *(const uint4*)fb;
                    b1.q = *(const uint4*)(fb + 128);
                    v16h B;
                    #pragma unroll
                    for (int i = 0; i < 8; ++i) { B[i] = b0.h[i]; B[8 + i] = b1.h[i]; }
                    if (w2 == 0)
                        accF[nt] = __builtin_amdgcn_wmma_f32_16x16x32_f16(
                            false, A, false, B, (short)0, accF[nt], false, false);
                    else
                        accS[nt] = __builtin_amdgcn_wmma_f32_16x16x32_f16(
                            false, A, false, B, (short)0, accS[nt], false, false);
                }
            }
        }

        // --- gate + scatter-add. C layout: lane<16 -> M=v, N=lane; lane>=16 -> M=v+8
        #pragma unroll
        for (int nt = 0; nt < NTILES; ++nt) {
            int   c  = nt * 16 + laneLo;
            float bf = bf_[c];
            float bs = bs_[c];
            #pragma unroll
            for (int v = 0; v < 8; ++v) {
                int   m   = v + (hiHalf ? 8 : 0);
                int   d   = ldsIdx[wv][m];
                float msg = fsig(accF[nt][v] + bf) * fsoftplus(accS[nt][v] + bs);
                unsafeAtomicAdd(agg + (long long)d * C_DIM + c, msg);
            }
        }
        __builtin_amdgcn_wave_barrier();
    }
}

// ---------------------------------------------------------------------------
// Node head: h = sigmoid(x+agg); o = sigmoid(h@W1+b1); pool += o
// ---------------------------------------------------------------------------
__global__ void node_kernel(const float* __restrict__ x,
                            const float* __restrict__ agg,
                            const long long* __restrict__ batch,
                            const float* __restrict__ W1,
                            const float* __restrict__ b1,
                            float* __restrict__ gpool, int N) {
    int n = blockIdx.x * blockDim.x + threadIdx.x;
    if (n >= N) return;
    float o[6];
    #pragma unroll
    for (int j = 0; j < 6; ++j) o[j] = b1[j];
    const float* xr = x   + (long long)n * C_DIM;
    const float* ar = agg + (long long)n * C_DIM;
    for (int c = 0; c < C_DIM; ++c) {
        float h = fsig(xr[c] + ar[c]);
        #pragma unroll
        for (int j = 0; j < 6; ++j) o[j] += h * W1[c * 6 + j];
    }
    int b = (int)batch[n];
    #pragma unroll
    for (int j = 0; j < 6; ++j)
        unsafeAtomicAdd(gpool + b * 6 + j, fsig(o[j]));
}

// ---------------------------------------------------------------------------
// Graph head: out = relu(g @ W2 + b2)
// ---------------------------------------------------------------------------
__global__ void head_kernel(const float* __restrict__ gpool,
                            const float* __restrict__ W2,
                            const float* __restrict__ b2,
                            float* __restrict__ out, int G) {
    int g = blockIdx.x * blockDim.x + threadIdx.x;
    if (g >= G) return;
    float acc = b2[0];
    #pragma unroll
    for (int j = 0; j < 6; ++j) acc += gpool[g * 6 + j] * W2[j];
    out[g] = acc > 0.0f ? acc : 0.0f;
}

extern "C" void kernel_launch(void* const* d_in, const int* in_sizes, int n_in,
                              void* d_out, int out_size, void* d_ws, size_t ws_size,
                              hipStream_t stream) {
    const float*     x     = (const float*)d_in[0];
    const long long* ei    = (const long long*)d_in[1];
    const float*     ea    = (const float*)d_in[2];
    const long long* batch = (const long long*)d_in[3];
    // d_in[4] = num_graphs (scalar) — G taken from out_size instead
    const float* Wf = (const float*)d_in[5];
    const float* bf = (const float*)d_in[6];
    const float* Ws = (const float*)d_in[7];
    const float* bs = (const float*)d_in[8];
    const float* W1 = (const float*)d_in[9];
    const float* b1 = (const float*)d_in[10];
    const float* W2 = (const float*)d_in[11];
    const float* b2 = (const float*)d_in[12];

    const int N = in_sizes[0] / C_DIM;
    const int E = in_sizes[2] / DE;
    const int G = out_size;

    char*     ws    = (char*)d_ws;
    uint32_t* wsB   = (uint32_t*)(ws + OFF_B);
    float*    gpool = (float*)(ws + OFF_GPOOL);
    float*    agg   = (float*)(ws + OFF_AGG);

    (void)hipMemsetAsync(agg,   0, (size_t)N * C_DIM * sizeof(float), stream);
    (void)hipMemsetAsync(gpool, 0, (size_t)G * 6 * sizeof(float), stream);

    prep_weights_kernel<<<(NFRAG * 256 + 255) / 256, 256, 0, stream>>>(Wf, Ws, wsB);

    const int numTiles = (E + 15) / 16;
    const int blocks   = (numTiles + WAVES * TPW - 1) / (WAVES * TPW);
    edge_kernel<<<blocks, 128, 0, stream>>>(x, ei, ea, bf, bs, wsB, agg, E, numTiles);

    node_kernel<<<(N + 255) / 256, 256, 0, stream>>>(x, agg, batch, W1, b1, gpool, N);
    head_kernel<<<(G + 255) / 256, 256, 0, stream>>>(gpool, W2, b2, (float*)d_out, G);
}
